// ViTSelfAttentionWithExtras_53738630808192
// MI455X (gfx1250) — matrix operand order repaired and checked by reading, not compiled, not run on hardware
//
#include <hip/hip_runtime.h>
#include <hip/hip_bf16.h>

typedef __attribute__((ext_vector_type(16))) _Float16 v16h;
typedef __attribute__((ext_vector_type(8)))  float    v8f;
typedef int i32x4 __attribute__((vector_size(16)));

#define H_HEADS 8
#define D_HEAD  64
#define N_SEQ   1024
#define B_BATCH 8
#define C_DIM   512
#define BN_TOK  (B_BATCH * N_SEQ)   // 8192 tokens

// ------------------------------------------------------------ async LDS copy

#if defined(__AMDGCN__) && __has_builtin(__builtin_amdgcn_global_load_async_to_lds_b128)
#define HAS_ASYNC_LDS 1
#else
#define HAS_ASYNC_LDS 0
#endif

#define AS1 __attribute__((address_space(1)))
#define AS3 __attribute__((address_space(3)))

// copy 16 bytes global -> LDS (per lane); async path uses ASYNCcnt
__device__ __forceinline__ void async_cp16(void* lds_dst, const void* gsrc) {
#if HAS_ASYNC_LDS
  __builtin_amdgcn_global_load_async_to_lds_b128(
      (AS1 i32x4*)(AS1 void*)(void*)gsrc,
      (AS3 i32x4*)(AS3 void*)lds_dst,
      /*offset=*/0, /*cpol=*/0);
#else
  *(uint4*)lds_dst = *(const uint4*)gsrc;
#endif
}

template <int N>
__device__ __forceinline__ void wait_async() {
#if HAS_ASYNC_LDS
#if __has_builtin(__builtin_amdgcn_s_wait_asynccnt)
  __builtin_amdgcn_s_wait_asynccnt(N);
#else
  asm volatile("s_wait_asynccnt %0" ::"i"(N));
#endif
#endif
}

// ---------------------------------------------------------------- WMMA utils

__device__ __forceinline__ v8f zero8() {
  v8f z = {0.f, 0.f, 0.f, 0.f, 0.f, 0.f, 0.f, 0.f};
  return z;
}

__device__ __forceinline__ v8f wmma_f16f32(v16h a, v16h b, v8f c) {
  return __builtin_amdgcn_wmma_f32_16x16x32_f16(
      /*neg_a=*/false, a, /*neg_b=*/false, b,
      /*c_mod=*/(short)0, c, /*reuse_a=*/false, /*reuse_b=*/false);
}

// A fragment, 16x32 f16 tile row-major in LDS with row stride `ld`.
// lanes 0-15: M=lane, K in {0..15}; lanes 16-31: M=lane-16, K in {16..31}
__device__ __forceinline__ v16h load_a_frag(const _Float16* base, int ld, int lane) {
  const int l  = lane & 15;
  const int kb = (lane < 16) ? 0 : 16;
  const _Float16* row = base + l * ld;
  v16h a;
#pragma unroll
  for (int h = 0; h < 16; ++h) a[h] = row[kb + h];
  return a;
}

// B fragment B[k][n], tile stored [n][k] (W row-major, or K-matrix for Q*K^T)
__device__ __forceinline__ v16h load_b_frag_nk(const _Float16* base, int ld, int lane) {
  const int col = lane & 15;
  const int kb  = (lane < 16) ? 0 : 16;
  const _Float16* row = base + col * ld;
  v16h b;
#pragma unroll
  for (int h = 0; h < 16; ++h) b[h] = row[kb + h];
  return b;
}

// B fragment B[k][n], tile stored [k][n] (V for P*V)
__device__ __forceinline__ v16h load_b_frag_kn(const _Float16* base, int ld, int lane) {
  const int col = lane & 15;
  const int kb  = (lane < 16) ? 0 : 16;
  v16h b;
#pragma unroll
  for (int h = 0; h < 16; ++h) b[h] = base[(kb + h) * ld + col];
  return b;
}

// ------------------------------------------------------------- f32->f16 pack

__global__ __launch_bounds__(256) void cvt_f32_f16(
    const float* __restrict__ src, _Float16* __restrict__ dst, int n4)
{
  const int i = blockIdx.x * 256 + threadIdx.x;
  if (i >= n4) return;
  const float4 f = ((const float4*)src)[i];
  union { _Float16 h[4]; uint2 u; } u;
  u.h[0] = (_Float16)f.x; u.h[1] = (_Float16)f.y;
  u.h[2] = (_Float16)f.z; u.h[3] = (_Float16)f.w;
  *(uint2*)(dst + (size_t)i * 4) = u.u;
}

// ------------------------------------------------------- generic GEMM kernel
// C[M,N] = A16[M,K] * W16[N,K]^T + bias[N]  (f16 operands, f32 accum/output)
// Double-buffered LDS tiles filled with async global->LDS copies.

#define GM  128
#define GN  128
#define GK  32
#define GLD (GK + 8)   // 40 halfs = 80B row pitch (16B multiple)

__global__ __launch_bounds__(256) void wmma_gemm_bias(
    const _Float16* __restrict__ A, const _Float16* __restrict__ W,
    const float* __restrict__ bias, float* __restrict__ Cout,
    int M, int N, int K)
{
  __shared__ _Float16 sA[2][GM][GLD];
  __shared__ _Float16 sB[2][GN][GLD];

  const int tid  = threadIdx.x;
  const int lane = tid & 31;
  const int wave = tid >> 5;     // 8 waves
  const int wm   = wave >> 2;    // 0..1 -> 64-row stripes
  const int wn   = wave & 3;     // 0..3 -> 32-col stripes
  const int m0   = blockIdx.y * GM;
  const int n0   = blockIdx.x * GN;

  v8f acc[4][2];
#pragma unroll
  for (int i = 0; i < 4; ++i)
#pragma unroll
    for (int j = 0; j < 2; ++j) acc[i][j] = zero8();

  // issue one 128x32 f16 tile (A and B): 512 chunks each, 2 per thread each
  auto issue_ab = [&](int k0, int st) {
#pragma unroll
    for (int it = 0; it < 2; ++it) {
      const int i = tid + it * 256;
      const int r = i >> 2, c = (i & 3) << 3;          // 8 halfs = 16B
      async_cp16(&sA[st][r][c], A + (size_t)(m0 + r) * K + k0 + c);
    }
#pragma unroll
    for (int it = 0; it < 2; ++it) {
      const int i = tid + it * 256;
      const int r = i >> 2, c = (i & 3) << 3;
      async_cp16(&sB[st][r][c], W + (size_t)(n0 + r) * K + k0 + c);
    }
  };

  issue_ab(0, 0);
  int stage = 0;
  for (int k0 = 0; k0 < K; k0 += GK, stage ^= 1) {
    __syncthreads();                       // other buffer free to overwrite
    if (k0 + GK < K) {
      issue_ab(k0 + GK, stage ^ 1);
      wait_async<4>();                     // current tile's 4 ops retired
    } else {
      wait_async<0>();
    }
    __syncthreads();                       // tile visible to all waves

    v16h bfr[2];
#pragma unroll
    for (int j = 0; j < 2; ++j)
      bfr[j] = load_b_frag_nk(&sB[stage][wn * 32 + j * 16][0], GLD, lane);
#pragma unroll
    for (int i = 0; i < 4; ++i) {
      v16h afr = load_a_frag(&sA[stage][wm * 64 + i * 16][0], GLD, lane);
#pragma unroll
      for (int j = 0; j < 2; ++j)
        acc[i][j] = wmma_f16f32(afr, bfr[j], acc[i][j]);
    }
  }

  // epilogue: C/D layout row = v + 8*(lane>=16), col = lane&15
  const int roff = (lane < 16) ? 0 : 8;
  const int col  = lane & 15;
  float* obase = Cout + (size_t)(m0 + wm * 64 + roff) * N + (n0 + wn * 32 + col);
#pragma unroll
  for (int i = 0; i < 4; ++i) {
#pragma unroll
    for (int j = 0; j < 2; ++j) {
      const float bv = bias[n0 + wn * 32 + j * 16 + col];
      float* p = obase + (size_t)i * 16 * N + j * 16;
#pragma unroll
      for (int v = 0; v < 8; ++v) p[(size_t)v * N] = acc[i][j][v] + bv;
    }
  }
}

// ----------------------------------------------- LayerNorm(q,k) + f16 packing
// qkv[token][3C] f32 -> q16/k16/v16 f16 in [B,H,N,D]; LN over D=64 for q,k.

__global__ __launch_bounds__(256) void ln_pack(
    const float* __restrict__ qkv,
    const float* __restrict__ qn_w, const float* __restrict__ qn_b,
    const float* __restrict__ kn_w, const float* __restrict__ kn_b,
    _Float16* __restrict__ q16, _Float16* __restrict__ k16,
    _Float16* __restrict__ v16_)
{
  const int lane = threadIdx.x & 31;
  const int gw   = (blockIdx.x * blockDim.x + threadIdx.x) >> 5; // wave per (token, head)
  if (gw >= BN_TOK * H_HEADS) return;
  const int token = gw >> 3;
  const int h     = gw & 7;
  const int b     = token >> 10;
  const int n     = token & 1023;
  const size_t src = (size_t)token * (3 * C_DIM) + h * D_HEAD;
  const size_t dst = (((size_t)(b * H_HEADS + h)) * N_SEQ + n) * D_HEAD;
  const int d0 = lane * 2;

  const float q0 = qkv[src + d0],             q1 = qkv[src + d0 + 1];
  const float k0 = qkv[src + C_DIM + d0],     k1 = qkv[src + C_DIM + d0 + 1];
  const float v0 = qkv[src + 2 * C_DIM + d0], v1 = qkv[src + 2 * C_DIM + d0 + 1];

  float qs = q0 + q1, qs2 = q0 * q0 + q1 * q1;
  float ks = k0 + k1, ks2 = k0 * k0 + k1 * k1;
#pragma unroll
  for (int m = 16; m > 0; m >>= 1) {
    qs += __shfl_xor(qs, m, 32);  qs2 += __shfl_xor(qs2, m, 32);
    ks += __shfl_xor(ks, m, 32);  ks2 += __shfl_xor(ks2, m, 32);
  }
  const float qmu = qs * (1.f / 64.f);
  const float qr  = rsqrtf(fmaxf(qs2 * (1.f / 64.f) - qmu * qmu, 0.f) + 1e-5f);
  const float kmu = ks * (1.f / 64.f);
  const float kr  = rsqrtf(fmaxf(ks2 * (1.f / 64.f) - kmu * kmu, 0.f) + 1e-5f);

  q16[dst + d0]     = (_Float16)((q0 - qmu) * qr * qn_w[d0]     + qn_b[d0]);
  q16[dst + d0 + 1] = (_Float16)((q1 - qmu) * qr * qn_w[d0 + 1] + qn_b[d0 + 1]);
  k16[dst + d0]     = (_Float16)((k0 - kmu) * kr * kn_w[d0]     + kn_b[d0]);
  k16[dst + d0 + 1] = (_Float16)((k1 - kmu) * kr * kn_w[d0 + 1] + kn_b[d0 + 1]);
  v16_[dst + d0]     = (_Float16)v0;
  v16_[dst + d0 + 1] = (_Float16)v1;
}

// --------------------------------------------------- flash attention + gate
// Per (b,h): O = softmax(Q K^T / 8) V, scaled by sigmoid(x_h . gate_w + gb).
// 128 q-rows/block, 8 waves (16 rows each). K/V tiles double-buffered via
// async global->LDS copies. ctx written as f16 for the projection GEMM.

#define BQ  128
#define BKV 128
#define LDH 72    // 144B row pitch (16B multiple)
#define LDP 136

__global__ __launch_bounds__(256) void attn_flash(
    const _Float16* __restrict__ q16, const _Float16* __restrict__ k16,
    const _Float16* __restrict__ v16_, const float* __restrict__ x,
    const float* __restrict__ gate_w, const float* __restrict__ gate_b,
    _Float16* __restrict__ ctx16)
{
  __shared__ _Float16 sQ[BQ][LDH];
  __shared__ _Float16 sK[2][BKV][LDH];
  __shared__ _Float16 sV[2][BKV][LDH];
  __shared__ _Float16 sP[8][16][LDP];   // per-wave P stripe (16 x 128)
  __shared__ float    sGate[BQ];

  const int tid  = threadIdx.x;
  const int lane = tid & 31;
  const int wave = tid >> 5;
  const int qblk = blockIdx.x;   // 0..7
  const int bh   = blockIdx.y;   // 0..63 (b*H + h)
  const int b    = bh >> 3;
  const int h    = bh & 7;
  const int n0   = qblk * BQ;

  const _Float16* qg = q16 + ((size_t)bh * N_SEQ + n0) * D_HEAD;
  // Q tile: 128x64 halfs = 1024 16B-chunks, 4 per thread
#pragma unroll
  for (int it = 0; it < 4; ++it) {
    const int i = tid + it * 256;
    const int r = i >> 3, c = (i & 7) << 3;
    async_cp16(&sQ[r][c], qg + (size_t)r * D_HEAD + c);
  }
  // issue one K/V tile pair into stage st (8 chunks per thread)
  auto issue_kv = [&](int kb, int st) {
    const _Float16* kg = k16 + ((size_t)bh * N_SEQ + kb * BKV) * D_HEAD;
    const _Float16* vg = v16_ + ((size_t)bh * N_SEQ + kb * BKV) * D_HEAD;
#pragma unroll
    for (int it = 0; it < 4; ++it) {
      const int i = tid + it * 256;
      const int r = i >> 3, c = (i & 7) << 3;
      async_cp16(&sK[st][r][c], kg + (size_t)r * D_HEAD + c);
    }
#pragma unroll
    for (int it = 0; it < 4; ++it) {
      const int i = tid + it * 256;
      const int r = i >> 3, c = (i & 7) << 3;
      async_cp16(&sV[st][r][c], vg + (size_t)r * D_HEAD + c);
    }
  };
  issue_kv(0, 0);

  // per-row conditional gate from x (f32)
  if (tid < BQ) {
    const int n = n0 + tid;
    const float* xp = x + ((size_t)(b * N_SEQ + n)) * C_DIM + h * D_HEAD;
    const float* gw = gate_w + h * D_HEAD;
    float a = gate_b[h];
#pragma unroll
    for (int d = 0; d < D_HEAD; d += 4) {
      const float4 xv = *(const float4*)(xp + d);
      const float4 wv = *(const float4*)(gw + d);
      a += xv.x * wv.x + xv.y * wv.y + xv.z * wv.z + xv.w * wv.w;
    }
    sGate[tid] = 1.f / (1.f + __expf(-a));
  }

  v8f   o[4];
  float mrow[8], lrow[8];
#pragma unroll
  for (int t = 0; t < 4; ++t) o[t] = zero8();
#pragma unroll
  for (int v = 0; v < 8; ++v) { mrow[v] = -1e30f; lrow[v] = 0.f; }

  const float scale = 0.125f;  // D^-0.5

  int stage = 0;
  for (int kb = 0; kb < N_SEQ / BKV; ++kb, stage ^= 1) {
    __syncthreads();                        // other K/V buffer free
    if (kb + 1 < N_SEQ / BKV) {
      issue_kv(kb + 1, stage ^ 1);
      wait_async<8>();                      // current tile (+Q on kb==0) done
    } else {
      wait_async<0>();
    }
    __syncthreads();

    // S (16 x 128) = Q_stripe (16x64) @ K^T (64x128)
    v8f s[8];
#pragma unroll
    for (int j = 0; j < 8; ++j) s[j] = zero8();
#pragma unroll
    for (int ks = 0; ks < 2; ++ks) {
      v16h a = load_a_frag(&sQ[wave * 16][ks * 32], LDH, lane);
#pragma unroll
      for (int j = 0; j < 8; ++j) {
        v16h bf = load_b_frag_nk(&sK[stage][j * 16][ks * 32], LDH, lane);
        s[j] = wmma_f16f32(a, bf, s[j]);
      }
    }

    // online softmax; rows live at (VGPR v, half-wave), cols across 16 lanes
    {
      float mt[8];
#pragma unroll
      for (int v = 0; v < 8; ++v) mt[v] = -1e30f;
#pragma unroll
      for (int j = 0; j < 8; ++j)
#pragma unroll
        for (int v = 0; v < 8; ++v) {
          s[j][v] *= scale;
          mt[v] = fmaxf(mt[v], s[j][v]);
        }
#pragma unroll
      for (int v = 0; v < 8; ++v) {
#pragma unroll
        for (int m = 8; m > 0; m >>= 1)
          mt[v] = fmaxf(mt[v], __shfl_xor(mt[v], m, 32));
        const float mn = fmaxf(mrow[v], mt[v]);
        const float rs = __expf(mrow[v] - mn);
        mrow[v] = mn;
        lrow[v] *= rs;
#pragma unroll
        for (int t = 0; t < 4; ++t) o[t][v] *= rs;
      }
      const int roff = (lane < 16) ? 0 : 8;
      const int col  = lane & 15;
      float rsum[8];
#pragma unroll
      for (int v = 0; v < 8; ++v) rsum[v] = 0.f;
#pragma unroll
      for (int j = 0; j < 8; ++j)
#pragma unroll
        for (int v = 0; v < 8; ++v) {
          const float p = __expf(s[j][v] - mrow[v]);
          rsum[v] += p;
          sP[wave][v + roff][j * 16 + col] = (_Float16)p;
        }
#pragma unroll
      for (int v = 0; v < 8; ++v) {
#pragma unroll
        for (int m = 8; m > 0; m >>= 1)
          rsum[v] += __shfl_xor(rsum[v], m, 32);
        lrow[v] += rsum[v];
      }
    }

    // O (16 x 64) += P (16x128) @ V (128x64)
#pragma unroll
    for (int ks = 0; ks < 4; ++ks) {
      v16h a = load_a_frag(&sP[wave][0][ks * 32], LDP, lane);
#pragma unroll
      for (int t = 0; t < 4; ++t) {
        v16h bf = load_b_frag_kn(&sV[stage][ks * 32][t * 16], LDH, lane);
        o[t] = wmma_f16f32(a, bf, o[t]);
      }
    }
  }

  // epilogue: normalize, gate, write f16 ctx (token-major [B*N, C])
  const int roff = (lane < 16) ? 0 : 8;
  const int col  = lane & 15;
#pragma unroll
  for (int v = 0; v < 8; ++v) {
    const int rloc = wave * 16 + v + roff;
    const int n    = n0 + rloc;
    const float g  = sGate[rloc] / lrow[v];
    _Float16* p = ctx16 + ((size_t)(b * N_SEQ + n)) * C_DIM + h * D_HEAD + col;
#pragma unroll
    for (int t = 0; t < 4; ++t) p[t * 16] = (_Float16)(o[t][v] * g);
  }
}

// ------------------------------------------------------------------- launch

extern "C" void kernel_launch(void* const* d_in, const int* in_sizes, int n_in,
                              void* d_out, int out_size, void* d_ws, size_t ws_size,
                              hipStream_t stream) {
  const float* x      = (const float*)d_in[0];
  const float* qkv_w  = (const float*)d_in[1];
  const float* qkv_b  = (const float*)d_in[2];
  const float* qn_w   = (const float*)d_in[3];
  const float* qn_b   = (const float*)d_in[4];
  const float* kn_w   = (const float*)d_in[5];
  const float* kn_b   = (const float*)d_in[6];
  const float* gate_w = (const float*)d_in[7];
  const float* gate_b = (const float*)d_in[8];
  const float* proj_w = (const float*)d_in[9];
  const float* proj_b = (const float*)d_in[10];
  float* out = (float*)d_out;

  // workspace layout (~90 MB)
  float*    qkv_ws = (float*)d_ws;                                    // 8192x1536 f32
  _Float16* q16    = (_Float16*)(qkv_ws + (size_t)BN_TOK * 3 * C_DIM);
  _Float16* k16    = q16 + (size_t)BN_TOK * C_DIM;
  _Float16* v16_   = k16 + (size_t)BN_TOK * C_DIM;
  _Float16* ctx16  = v16_ + (size_t)BN_TOK * C_DIM;                   // 8192x512 f16
  _Float16* x16    = ctx16 + (size_t)BN_TOK * C_DIM;                  // 8192x512 f16
  _Float16* w16q   = x16 + (size_t)BN_TOK * C_DIM;                    // 1536x512 f16
  _Float16* w16p   = w16q + (size_t)3 * C_DIM * C_DIM;                // 512x512 f16

  // 0. pack f32 -> f16 (x, qkv_w, proj_w)
  {
    const int nx = BN_TOK * C_DIM / 4, nq = 3 * C_DIM * C_DIM / 4,
              np = C_DIM * C_DIM / 4;
    cvt_f32_f16<<<dim3((nx + 255) / 256), 256, 0, stream>>>(x, x16, nx);
    cvt_f32_f16<<<dim3((nq + 255) / 256), 256, 0, stream>>>(qkv_w, w16q, nq);
    cvt_f32_f16<<<dim3((np + 255) / 256), 256, 0, stream>>>(proj_w, w16p, np);
  }
  // 1. QKV GEMM: [8192,512]f16 x [1536,512]^T f16 -> f32
  wmma_gemm_bias<<<dim3((3 * C_DIM) / GN, BN_TOK / GM), 256, 0, stream>>>(
      x16, w16q, qkv_b, qkv_ws, BN_TOK, 3 * C_DIM, C_DIM);
  // 2. LayerNorm q,k + pack f16 [B,H,N,D]
  ln_pack<<<dim3((BN_TOK * H_HEADS * 32) / 256), 256, 0, stream>>>(
      qkv_ws, qn_w, qn_b, kn_w, kn_b, q16, k16, v16_);
  // 3. flash attention + gate, per (b,h); ctx in f16
  attn_flash<<<dim3(N_SEQ / BQ, B_BATCH * H_HEADS), 256, 0, stream>>>(
      q16, k16, v16_, x, gate_w, gate_b, ctx16);
  // 4. output projection: [8192,512]f16 x [512,512]^T f16 -> f32 out
  wmma_gemm_bias<<<dim3(C_DIM / GN, BN_TOK / GM), 256, 0, stream>>>(
      ctx16, w16p, proj_b, out, BN_TOK, C_DIM, C_DIM);
}
